// NystromAttention_22771916603595
// MI455X (gfx1250) — compile-verified, matrix operand order checked by reading, not run on hardware
//
#include <hip/hip_runtime.h>
#include <hip/hip_bf16.h>

typedef __bf16 bf16_t;
typedef __attribute__((ext_vector_type(16))) __bf16 bfrag_t;
typedef __attribute__((ext_vector_type(8)))  __bf16 bh8_t;   // 16-byte packed
typedef __attribute__((ext_vector_type(8)))  float  v8f_t;
typedef __attribute__((ext_vector_type(4)))  unsigned int u32x4;
typedef __attribute__((ext_vector_type(8)))  unsigned int u32x8;

// Problem constants: B=4,H=8 -> BH=32; N=4096; D=64; M=256; L=16; Kconv=33
#define BH    32
#define NTOK  4096
#define DDIM  64
#define MLM   256
#define KCONV 33

// ---------------------------------------------------------------------------
// WMMA fragment helpers (CDNA5 ISA 7.12.2, wave32).
// bf16 A/B fragment element e -> K = kh*8+e (e<8) or 16+kh*8+(e-8) (e>=8):
// per lane the fragment is TWO CONTIGUOUS 16B chunks of a K-contiguous source.
// ---------------------------------------------------------------------------
__device__ __forceinline__ bfrag_t load_frag_kc(const bf16_t* p, long sL) {
  int lane = threadIdx.x & 31;
  int li = lane & 15;
  int kh = (lane >> 4) & 1;
  const bf16_t* base = p + (long)li * sL + kh * 8;
  bh8_t lo = *(const bh8_t*)(base);        // K = kh*8 .. kh*8+7
  bh8_t hi = *(const bh8_t*)(base + 16);   // K = 16+kh*8 .. 16+kh*8+7
  return __builtin_shufflevector(lo, hi, 0, 1, 2, 3, 4, 5, 6, 7, 8, 9, 10, 11,
                                 12, 13, 14, 15);
}

__device__ __forceinline__ v8f_t wmma_bf16(bfrag_t a, bfrag_t b, v8f_t c) {
  return __builtin_amdgcn_wmma_f32_16x16x32_bf16(false, a, false, b, (short)0, c,
                                                 false, false);
}

// LDS byte offset of a __shared__ object (generic -> AS(3) -> 32-bit offset)
__device__ __forceinline__ unsigned lds_off_u32(const void* p) {
  return (unsigned)(unsigned long long)(
      __attribute__((address_space(3))) const char*)p;
}

// ---------------------------------------------------------------------------
// Tensor Data Mover: async DMA of a 2D bf16 tile [tileH x tileW] from global
// (row stride strideElems) into LDS at lds_byte_off, padding each 64B row
// with 16B (pad_interval=16DW, pad_amount=4DW -> 40-element LDS rows).
// D# layout per CDNA5 ISA 8.3/8.4; tracked by TENSORcnt.
// ---------------------------------------------------------------------------
__device__ __forceinline__ void tdm_load_2d_to_lds(const void* gptr,
                                                   unsigned lds_byte_off,
                                                   unsigned tileW, unsigned tileH,
                                                   unsigned strideElems) {
  unsigned long long ga = (unsigned long long)gptr;
  u32x4 g0;
  g0[0] = 1u;                               // count=1 (valid user descriptor)
  g0[1] = lds_byte_off;                     // lds_addr (bytes)
  g0[2] = (unsigned)ga;                     // global_addr[31:0]
  g0[3] = (unsigned)((ga >> 32) & 0x01FFFFFFu) | (2u << 30);  // [56:32]|type=2
  u32x8 g1;
  g1[0] = (1u << 16)                        // data_size: 2 bytes
        | (1u << 20) | (3u << 22) | (3u << 25);  // pad_en, 16DW interval, 4DW
  unsigned dim0 = 1u << 20, dim1 = 1u << 20;     // generous OOB bounds
  g1[1] = (dim0 & 0xFFFFu) << 16;           // tensor_dim0 lo16 (bits 63:48)
  g1[2] = (dim0 >> 16) | ((dim1 & 0xFFFFu) << 16);  // dim0 hi | dim1 lo
  g1[3] = (dim1 >> 16) | (tileW << 16);     // dim1 hi | tile_dim0
  g1[4] = tileH;                            // tile_dim1 (tile_dim2 = 0)
  g1[5] = strideElems;                      // tensor_dim0_stride lo32
  g1[6] = 0u;
  g1[7] = 0u;
  asm volatile("tensor_load_to_lds %0, %1" ::"s"(g0), "s"(g1) : "memory");
}

// ---------------------------------------------------------------------------
// Elementwise prep: qb = bf16(q*0.125) row-major, kb = bf16(k) row-major
// ---------------------------------------------------------------------------
__global__ __launch_bounds__(256) void prep_kernel(
    const float* __restrict__ q, const float* __restrict__ k,
    bf16_t* __restrict__ qb, bf16_t* __restrict__ kb, long total) {
  long idx = (long)blockIdx.x * blockDim.x + threadIdx.x;
  if (idx >= total) return;
  qb[idx] = (bf16_t)(q[idx] * 0.125f);
  kb[idx] = (bf16_t)k[idx];
}

// vT[bh][d][n] = bf16(v[bh][n][d])  (output-coalesced transpose)
__global__ __launch_bounds__(256) void vt_kernel(const float* __restrict__ v,
                                                 bf16_t* __restrict__ vt,
                                                 long total) {
  long idx = (long)blockIdx.x * blockDim.x + threadIdx.x;
  if (idx >= total) return;
  int n = (int)(idx & 4095);
  int d = (int)((idx >> 12) & 63);
  long bh = idx >> 18;
  vt[idx] = (bf16_t)v[(bh << 18) + (long)n * 64 + d];
}

// Landmarks: mean over groups of 16 tokens (q side includes 0.125 scale)
__global__ __launch_bounds__(256) void landmark_kernel(
    const float* __restrict__ q, const float* __restrict__ k,
    bf16_t* __restrict__ qlb, bf16_t* __restrict__ klb, long total) {
  long idx = (long)blockIdx.x * blockDim.x + threadIdx.x;
  if (idx >= total) return;  // total = BH*M*D
  int d = (int)(idx & 63);
  int m = (int)((idx >> 6) & 255);
  long bh = idx >> 14;
  const float* qp = q + (bh << 18) + (long)m * 1024 + d;
  const float* kp = k + (bh << 18) + (long)m * 1024 + d;
  float sq = 0.f, sk = 0.f;
#pragma unroll
  for (int l = 0; l < 16; ++l) { sq += qp[l * 64]; sk += kp[l * 64]; }
  qlb[idx] = (bf16_t)(sq * (0.125f / 16.0f));
  klb[idx] = (bf16_t)(sk * (1.0f / 16.0f));
}

// ---------------------------------------------------------------------------
// Batched WMMA GEMM, NT form: P = A[256xK] * B^T, B stored [N][K].
// Block = 8 waves; wave w owns rows 32w..32w+31 (2 m-tiles) -> M=256/block.
// Blocks split the N range (nPerBlock). The B panel [64 x 32] is DMA'd into
// LDS by the Tensor Data Mover, double-buffered: wave 0 issues the TDM for
// chunk s+1 into the alternate buffer, waits TENSORcnt<=1, then all waves
// compute chunk s (4 global b128 + 8 ds b128 per 8 WMMAs).
// Epilogues (diag c applies at gm==gn; transposed stores are packed b128):
//   0: outF(rm f32) = P
//   1: out0(rm) = 7I - P ; out1(T) = P            (pinv G1: t, xzT)
//   2: out0(rm) = cI - P                          (pinv G2: t2)
//   3: out0(T)  = cI - P                          (pinv G3: t3T)
//   4: out0(rm) = 0.25P ; out1(T) = 0.25P         (pinv G4: z, zT)
//   5: out0(T)  = P                               (mid2T)
// ---------------------------------------------------------------------------
__global__ __launch_bounds__(256) void gemm_bf_kernel(
    const bf16_t* __restrict__ A, const bf16_t* __restrict__ B,
    float* __restrict__ outF, bf16_t* __restrict__ out0,
    bf16_t* __restrict__ out1, int M, int N, int K, int lda, int ldb, int ldc,
    long bsA, long bsB, long bsC, int nPerBlock, int mode, float cdiag) {
  __shared__ bf16_t lds_b[2][64 * 40];  // double-buffered 64x32 panel, 80B rows

  int batch = blockIdx.y;
  int tid = threadIdx.x;
  int wave = tid >> 5;
  int lane = tid & 31;
  int nl = lane & 15;
  int roff = (lane >> 4) * 8;
  int m0base = wave * 32;  // two 16-row tiles per wave
  const bf16_t* Ab = A + bsA * batch;
  const bf16_t* Bb = B + bsB * batch;
  unsigned ldsOff0 = lds_off_u32(&lds_b[0][0]);
  unsigned ldsOff1 = lds_off_u32(&lds_b[1][0]);

  int nEnd = blockIdx.x * nPerBlock + nPerBlock;
  if (nEnd > N) nEnd = N;
  for (int n0 = blockIdx.x * nPerBlock; n0 < nEnd; n0 += 64) {
    v8f_t acc[2][4];
#pragma unroll
    for (int mt = 0; mt < 2; ++mt)
#pragma unroll
      for (int t = 0; t < 4; ++t) { v8f_t z = {}; acc[mt][t] = z; }

    int steps = K >> 5;  // always even here (K = 64 or 256)
    if (wave == 0)       // prefetch chunk 0
      tdm_load_2d_to_lds(Bb + (long)n0 * ldb, ldsOff0, 32, 64, (unsigned)ldb);

    for (int s = 0; s < steps; ++s) {
      int k0 = s << 5;
      if (wave == 0) {
        if (s + 1 < steps) {
          tdm_load_2d_to_lds(Bb + (long)n0 * ldb + k0 + 32,
                             ((s + 1) & 1) ? ldsOff1 : ldsOff0, 32, 64,
                             (unsigned)ldb);
          __builtin_amdgcn_s_wait_tensorcnt(1);  // chunk s landed, s+1 in flight
        } else {
          __builtin_amdgcn_s_wait_tensorcnt(0);
        }
      }
      __syncthreads();  // chunk s visible to all waves
      const bf16_t* bbuf = &lds_b[s & 1][0];
      bfrag_t a0 = load_frag_kc(Ab + (long)m0base * lda + k0, lda);
      bfrag_t a1 = load_frag_kc(Ab + (long)(m0base + 16) * lda + k0, lda);
#pragma unroll
      for (int t = 0; t < 4; ++t) {
        bfrag_t b = load_frag_kc(bbuf + t * 16 * 40, 40);
        acc[0][t] = wmma_bf16(a0, b, acc[0][t]);
        acc[1][t] = wmma_bf16(a1, b, acc[1][t]);
      }
      __syncthreads();  // reads of this buffer done before TDM rewrites it
    }
    // ---- epilogue ----
#pragma unroll
    for (int mt = 0; mt < 2; ++mt) {
      int m0 = m0base + mt * 16;
#pragma unroll
      for (int t = 0; t < 4; ++t) {
        int gn = n0 + 16 * t + nl;
        long rmBase = bsC * batch;
        if (mode == 0) {
#pragma unroll
          for (int r = 0; r < 8; ++r)
            outF[rmBase + (long)(m0 + roff + r) * ldc + gn] = acc[mt][t][r];
        } else if (mode == 1) {
          bh8_t pk;
#pragma unroll
          for (int r = 0; r < 8; ++r) {
            int gm = m0 + roff + r;
            float p = acc[mt][t][r];
            out0[rmBase + (long)gm * ldc + gn] =
                (bf16_t)(((gm == gn) ? 7.0f : 0.0f) - p);
            pk[r] = (bf16_t)p;
          }
          *(bh8_t*)(out1 + rmBase + (long)gn * M + m0 + roff) = pk;
        } else if (mode == 2) {
#pragma unroll
          for (int r = 0; r < 8; ++r) {
            int gm = m0 + roff + r;
            out0[rmBase + (long)gm * ldc + gn] =
                (bf16_t)(((gm == gn) ? cdiag : 0.0f) - acc[mt][t][r]);
          }
        } else if (mode == 3) {
          bh8_t pk;
#pragma unroll
          for (int r = 0; r < 8; ++r) {
            int gm = m0 + roff + r;
            pk[r] = (bf16_t)(((gm == gn) ? cdiag : 0.0f) - acc[mt][t][r]);
          }
          *(bh8_t*)(out0 + rmBase + (long)gn * M + m0 + roff) = pk;
        } else if (mode == 4) {
          bh8_t pk;
#pragma unroll
          for (int r = 0; r < 8; ++r) {
            int gm = m0 + roff + r;
            float p = 0.25f * acc[mt][t][r];
            out0[rmBase + (long)gm * ldc + gn] = (bf16_t)p;
            pk[r] = (bf16_t)p;
          }
          *(bh8_t*)(out1 + rmBase + (long)gn * M + m0 + roff) = pk;
        } else {  // 5
          bh8_t pk;
#pragma unroll
          for (int r = 0; r < 8; ++r) pk[r] = (bf16_t)acc[mt][t][r];
          *(bh8_t*)(out0 + rmBase + (long)gn * M + m0 + roff) = pk;
        }
      }
    }
  }
}

// ---------------------------------------------------------------------------
// Row softmax over 256-wide rows (attn2)
// ---------------------------------------------------------------------------
__global__ __launch_bounds__(256) void softmax256_kernel(float* __restrict__ x) {
  __shared__ float red[256];
  long row = blockIdx.x;
  int t = threadIdx.x;
  float v = x[row * 256 + t];
  red[t] = v;
  __syncthreads();
  for (int s = 128; s > 0; s >>= 1) {
    if (t < s) red[t] = fmaxf(red[t], red[t + s]);
    __syncthreads();
  }
  float m = red[0];
  __syncthreads();
  float e = __expf(v - m);
  red[t] = e;
  __syncthreads();
  for (int s = 128; s > 0; s >>= 1) {
    if (t < s) red[t] += red[t + s];
    __syncthreads();
  }
  x[row * 256 + t] = e / red[0];
}

__global__ void init_scalar_kernel(float* p) { *p = 0.0f; }

// max over all (bh) of column sums of attn2 (row sums are exactly 1)
__global__ __launch_bounds__(256) void colmax_kernel(
    const float* __restrict__ attn2, float* __restrict__ maxsc) {
  __shared__ float red[256];
  int bh = blockIdx.x, j = threadIdx.x;
  const float* x = attn2 + (long)bh * 65536;
  float s = 0.f;
  for (int i = 0; i < 256; ++i) s += x[i * 256 + j];
  red[j] = s;
  __syncthreads();
  for (int st = 128; st > 0; st >>= 1) {
    if (j < st) red[j] = fmaxf(red[j], red[j + st]);
    __syncthreads();
  }
  if (j == 0) atomicMax((int*)maxsc, __float_as_int(red[0]));
}

// xb = bf16(attn2) rm; z0 = attn2^T/maxcs rm; z0T = attn2/maxcs (contiguous)
__global__ __launch_bounds__(256) void zprep_kernel(
    const float* __restrict__ attn2, const float* __restrict__ maxsc,
    bf16_t* __restrict__ xb, bf16_t* __restrict__ zrm,
    bf16_t* __restrict__ zT, long total) {
  long idx = (long)blockIdx.x * blockDim.x + threadIdx.x;
  if (idx >= total) return;
  float inv = 1.0f / *maxsc;
  long bh = idx >> 16;
  int ij = (int)(idx & 65535);
  int i = ij >> 8, j = ij & 255;
  float xv = attn2[idx];
  xb[idx] = (bf16_t)xv;
  zT[idx] = (bf16_t)(xv * inv);
  zrm[idx] = (bf16_t)(attn2[(bh << 16) | (j << 8) | i] * inv);
}

// ---------------------------------------------------------------------------
// Fused  out = softmax(Q·K^T) · V  (flash-style, all WMMA).
// Q:[rows x 64] K-contig; Km:[cols x 64] K-contig (= right-factor transposed);
// VT:[64 x cols] so V fragments are K-contiguous b128 pairs.
// outF: row-major f32 [rows x 64]; outBT: transposed bf16 [64 x rows] (b128).
// ---------------------------------------------------------------------------
__global__ __launch_bounds__(256) void fused_softmax_av_kernel(
    const bf16_t* __restrict__ Q, const bf16_t* __restrict__ Km,
    const bf16_t* __restrict__ VT, float* __restrict__ outF,
    bf16_t* __restrict__ outBT, int rows, int cols, long bsQ, long bsK,
    long bsV, long bsO) {
  __shared__ float  lds_s[8][16][33];
  __shared__ bf16_t lds_p[8][16][40];  // 80B row stride: 16B aligned for b128
  __shared__ float  lds_f[8][16];

  int batch = blockIdx.y;
  int w = threadIdx.x >> 5;
  int lane = threadIdx.x & 31;
  int nl = lane & 15;
  int roff = (lane >> 4) * 8;
  int m0 = (blockIdx.x * 8 + w) * 16;  // grid sized so every wave is live

  const bf16_t* Qb = Q + bsQ * batch;
  const bf16_t* Kb = Km + bsK * batch;
  const bf16_t* Vb = VT + bsV * batch;

  v8f_t acc[4];
#pragma unroll
  for (int t = 0; t < 4; ++t) { v8f_t z = {}; acc[t] = z; }
  float run_max = -3.0e38f, run_sum = 0.0f;

  bfrag_t aq0 = load_frag_kc(Qb + (long)m0 * 64 + 0, 64);
  bfrag_t aq1 = load_frag_kc(Qb + (long)m0 * 64 + 32, 64);

  for (int j0 = 0; j0 < cols; j0 += 32) {
    // ---- S[16x32] = Q_tile · K^T (hoist all K frags, then 4 WMMAs) ----
    bfrag_t bk[4];
#pragma unroll
    for (int s = 0; s < 2; ++s) {
      bk[2 * s + 0] = load_frag_kc(Kb + (long)(j0 + s * 16) * 64 + 0, 64);
      bk[2 * s + 1] = load_frag_kc(Kb + (long)(j0 + s * 16) * 64 + 32, 64);
    }
#pragma unroll
    for (int s = 0; s < 2; ++s) {
      v8f_t c = {};
      c = wmma_bf16(aq0, bk[2 * s + 0], c);
      c = wmma_bf16(aq1, bk[2 * s + 1], c);
#pragma unroll
      for (int r = 0; r < 8; ++r) lds_s[w][roff + r][s * 16 + nl] = c[r];
    }
    __syncthreads();
    // ---- online softmax stats; lanes 0..15 own one row each ----
    if (lane < 16) {
      float vals[32];
      float mloc = run_max;
#pragma unroll
      for (int j = 0; j < 32; ++j) {
        vals[j] = lds_s[w][lane][j];
        mloc = fmaxf(mloc, vals[j]);
      }
      float corr = __expf(run_max - mloc);
      float ssum = 0.f;
#pragma unroll
      for (int j = 0; j < 32; ++j) {
        float e = __expf(vals[j] - mloc);
        ssum += e;
        lds_p[w][lane][j] = (bf16_t)e;
      }
      run_sum = run_sum * corr + ssum;
      run_max = mloc;
      lds_f[w][lane] = corr;
    }
    __syncthreads();
    // ---- rescale accumulator, then acc += P · V_chunk ----
    bfrag_t bv[4];
#pragma unroll
    for (int t = 0; t < 4; ++t)
      bv[t] = load_frag_kc(Vb + (long)(t * 16) * cols + j0, cols);
#pragma unroll
    for (int t = 0; t < 4; ++t)
#pragma unroll
      for (int r = 0; r < 8; ++r) acc[t][r] *= lds_f[w][roff + r];
    bfrag_t ap = load_frag_kc(&lds_p[w][0][0], 40);
#pragma unroll
    for (int t = 0; t < 4; ++t) acc[t] = wmma_bf16(ap, bv[t], acc[t]);
    __syncthreads();
  }
  // ---- normalize and write ----
  if (lane < 16) lds_f[w][lane] = 1.0f / run_sum;
  __syncthreads();
#pragma unroll
  for (int t = 0; t < 4; ++t) {
    int gn = t * 16 + nl;
    float o[8];
#pragma unroll
    for (int r = 0; r < 8; ++r) o[r] = acc[t][r] * lds_f[w][roff + r];
    if (outF) {
#pragma unroll
      for (int r = 0; r < 8; ++r)
        outF[bsO * batch + (long)(m0 + roff + r) * 64 + gn] = o[r];
    }
    if (outBT) {
      bh8_t pk;
#pragma unroll
      for (int r = 0; r < 8; ++r) pk[r] = (bf16_t)o[r];
      *(bh8_t*)(outBT + bsO * batch + (long)gn * rows + m0 + roff) = pk;
    }
  }
}

// ---------------------------------------------------------------------------
// Depthwise residual conv over tokens (K=33, pad 16): out += conv(v)
// ---------------------------------------------------------------------------
__global__ __launch_bounds__(256) void conv_res_kernel(
    const float* __restrict__ v, const float* __restrict__ w,
    float* __restrict__ out, long total) {
  long idx = (long)blockIdx.x * blockDim.x + threadIdx.x;
  if (idx >= total) return;
  int d = (int)(idx & 63);
  int n = (int)((idx >> 6) & 4095);
  long bh = idx >> 18;
  int h = (int)(bh & 7);
  const float* vp = v + (bh << 18) + d;
  float s = 0.f;
#pragma unroll
  for (int t = 0; t < KCONV; ++t) {
    int nn = n + t - 16;
    if (nn >= 0 && nn < NTOK) s += vp[(long)nn * 64] * w[h * KCONV + t];
  }
  out[idx] += s;
}

// ---------------------------------------------------------------------------
extern "C" void kernel_launch(void* const* d_in, const int* in_sizes, int n_in,
                              void* d_out, int out_size, void* d_ws,
                              size_t ws_size, hipStream_t stream) {
  const float* q = (const float*)d_in[0];
  const float* k = (const float*)d_in[1];
  const float* v = (const float*)d_in[2];
  const float* rw = (const float*)d_in[3];
  float* out = (float*)d_out;

  char* ws = (char*)d_ws;
  size_t off = 0;
  auto alloc = [&](size_t bytes) {
    void* p = ws + off;
    off += (bytes + 255) & ~(size_t)255;
    return p;
  };
  const long nqkv = (long)BH * NTOK * DDIM;  // 8388608
  const long nlm = (long)BH * MLM * DDIM;    // 524288
  const long nm2 = (long)BH * MLM * MLM;     // 2097152

  bf16_t* qb   = (bf16_t*)alloc(nqkv * 2);   // qs, row-major [N][64]
  bf16_t* kb   = (bf16_t*)alloc(nqkv * 2);   // k,  row-major [N][64]
  bf16_t* vT   = (bf16_t*)alloc(nqkv * 2);   // v^T [64][N]
  bf16_t* qlb  = (bf16_t*)alloc(nlm * 2);
  bf16_t* klb  = (bf16_t*)alloc(nlm * 2);
  float*  attn2 = (float*)alloc(nm2 * 4);
  float*  maxsc = (float*)alloc(256);
  bf16_t* xb   = (bf16_t*)alloc(nm2 * 2);    // attn2 rm (A operand)
  bf16_t* z0rm = (bf16_t*)alloc(nm2 * 2);
  bf16_t* z0T  = (bf16_t*)alloc(nm2 * 2);
  bf16_t* z1rm = (bf16_t*)alloc(nm2 * 2);
  bf16_t* z1T  = (bf16_t*)alloc(nm2 * 2);
  bf16_t* xzT  = (bf16_t*)alloc(nm2 * 2);
  bf16_t* tA   = (bf16_t*)alloc(nm2 * 2);    // t  (rm)
  bf16_t* tB   = (bf16_t*)alloc(nm2 * 2);    // t2 (rm)
  bf16_t* t3T  = (bf16_t*)alloc(nm2 * 2);    // t3 transposed
  bf16_t* mid1T = (bf16_t*)alloc(nlm * 2);   // [64][256]
  bf16_t* mid2T = (bf16_t*)alloc(nlm * 2);   // [64][256]

  // 1) bf16 conversions, v transpose, landmarks
  prep_kernel<<<(int)(nqkv / 256), 256, 0, stream>>>(q, k, qb, kb, nqkv);
  vt_kernel<<<(int)(nqkv / 256), 256, 0, stream>>>(v, vT, nqkv);
  landmark_kernel<<<(int)(nlm / 256), 256, 0, stream>>>(q, k, qlb, klb, nlm);

  // 2) sim2 = q_l · k_l^T (f32) -> row softmax -> attn2
  gemm_bf_kernel<<<dim3(2, BH), 256, 0, stream>>>(
      qlb, klb, attn2, nullptr, nullptr, MLM, MLM, DDIM, 64, 64, 256, 16384,
      16384, 65536, 128, 0, 0.f);
  softmax256_kernel<<<BH * MLM, 256, 0, stream>>>(attn2);

  // 3) pinv init: z0 = attn2^T / max(colsum)
  init_scalar_kernel<<<1, 1, 0, stream>>>(maxsc);
  colmax_kernel<<<BH, 256, 0, stream>>>(attn2, maxsc);
  zprep_kernel<<<(int)(nm2 / 256), 256, 0, stream>>>(attn2, maxsc, xb, z0rm,
                                                     z0T, nm2);

  // 4) Newton-Schulz x6. Polynomials in xz commute, so xz is only ever a
  //    (transposed) B operand; t/t2 stay row-major A operands.
  bf16_t *zcr = z0rm, *zcT = z0T, *znr = z1rm, *znT = z1T;
  for (int it = 0; it < 6; ++it) {
    gemm_bf_kernel<<<dim3(2, BH), 256, 0, stream>>>(  // t = 7I - x·z ; xzT
        xb, zcT, nullptr, tA, xzT, MLM, MLM, MLM, 256, 256, 256, 65536, 65536,
        65536, 128, 1, 0.f);
    gemm_bf_kernel<<<dim3(2, BH), 256, 0, stream>>>(  // t2 = 15I - t·xz
        tA, xzT, nullptr, tB, nullptr, MLM, MLM, MLM, 256, 256, 256, 65536,
        65536, 65536, 128, 2, 15.f);
    gemm_bf_kernel<<<dim3(2, BH), 256, 0, stream>>>(  // t3T = (13I - t2·xz)^T
        tB, xzT, nullptr, t3T, nullptr, MLM, MLM, MLM, 256, 256, 256, 65536,
        65536, 65536, 128, 3, 13.f);
    gemm_bf_kernel<<<dim3(2, BH), 256, 0, stream>>>(  // z' = 0.25 z·t3 (rm+T)
        zcr, t3T, nullptr, znr, znT, MLM, MLM, MLM, 256, 256, 256, 65536,
        65536, 65536, 128, 4, 0.f);
    bf16_t* t;
    t = zcr; zcr = znr; znr = t;
    t = zcT; zcT = znT; znT = t;
  }

  // 5) mid1^T = [softmax(q_l·k^T) · v]^T   [64][256]
  fused_softmax_av_kernel<<<dim3(2, BH), 256, 0, stream>>>(
      qlb, kb, vT, nullptr, mid1T, MLM, NTOK, 16384, 262144, 262144, 16384);

  // 6) mid2^T = (z · mid1)^T               [64][256]
  gemm_bf_kernel<<<dim3(1, BH), 256, 0, stream>>>(
      zcr, mid1T, nullptr, mid2T, nullptr, MLM, DDIM, MLM, 256, 256, 64, 65536,
      16384, 16384, 64, 5, 0.f);

  // 7) out = softmax(qs·k_l^T) · mid2      [BH,4096,64] f32 -> d_out
  fused_softmax_av_kernel<<<dim3(32, BH), 256, 0, stream>>>(
      qb, klb, mid2T, out, nullptr, NTOK, MLM, 262144, 16384, 16384, 262144);

  // 8) out += depthwise residual conv
  conv_res_kernel<<<(int)(nqkv / 256), 256, 0, stream>>>(v, rw, out, nqkv);
}
